// SelfAttention_44203803410623
// MI455X (gfx1250) — matrix operand order repaired and checked by reading, not compiled, not run on hardware
//
#include <hip/hip_runtime.h>
#include <hip/hip_bf16.h>

#define D_MODEL  1024
#define N_HEADS  16
#define HEAD_DIM 64
#define BATCH    2
#define SEQ      2048

typedef __attribute__((ext_vector_type(16))) _Float16 v16h;
typedef __attribute__((ext_vector_type(8)))  _Float16 v8h;
typedef __attribute__((ext_vector_type(8)))  float    v8f;
typedef __attribute__((ext_vector_type(4)))  float    v4f;
typedef __attribute__((ext_vector_type(4)))  unsigned int u32x4;
typedef __attribute__((ext_vector_type(8)))  int      i32x8;
typedef __attribute__((ext_vector_type(4)))  int      i32x4;

#if __has_builtin(__builtin_amdgcn_tensor_load_to_lds)
#define HAVE_TDM 1
#else
#define HAVE_TDM 0
#endif

union V16H { v16h v; v8h h[2]; };

struct Raw16 { v4f q[4]; };   // 16 raw fp32 values staged for conversion

static __device__ __forceinline__ void ldA_raw(const float* __restrict__ row,
                                               int k0, int koffA, Raw16& r) {
  const float* p0 = row + k0 + koffA;
  const float* p1 = row + k0 + 16 + koffA;
  r.q[0] = *(const v4f*)p0;  r.q[1] = *(const v4f*)(p0 + 4);
  r.q[2] = *(const v4f*)p1;  r.q[3] = *(const v4f*)(p1 + 4);
}

static __device__ __forceinline__ void ldB_raw(const float* __restrict__ p, Raw16& r) {
  r.q[0] = *(const v4f*)p;        r.q[1] = *(const v4f*)(p + 4);
  r.q[2] = *(const v4f*)(p + 16);  r.q[3] = *(const v4f*)(p + 20);
}

static __device__ __forceinline__ v16h cvt_frag(const Raw16& r) {
  union { v8f f; v4f q[2]; } lo, hi;
  lo.q[0] = r.q[0]; lo.q[1] = r.q[1];
  hi.q[0] = r.q[2]; hi.q[1] = r.q[3];
  V16H u;
  u.h[0] = __builtin_convertvector(lo.f, v8h);
  u.h[1] = __builtin_convertvector(hi.f, v8h);
  return u.v;
}

static __device__ __forceinline__ v16h loadA_f16(const _Float16* row, int koff) {
  V16H u;
  u.h[0] = *(const v8h*)(row + koff);
  u.h[1] = *(const v8h*)(row + 16 + koff);
  return u.v;
}

static __device__ __forceinline__ v16h loadB_f16(const _Float16* p) {
  V16H u;
  u.h[0] = *(const v8h*)(p);
  u.h[1] = *(const v8h*)(p + 16);
  return u.v;
}

static __device__ __forceinline__ v8f wmma_f16(v16h a, v16h b, v8f c) {
  return __builtin_amdgcn_wmma_f32_16x16x32_f16(false, a, false, b,
                                                (short)0, c, false, false);
}

// Low 32 bits of a flat LDS pointer = byte offset in the group's LDS segment.
static __device__ __forceinline__ unsigned int lds_off(const void* p) {
  return (unsigned int)(unsigned long long)p;
}

// ---------------------------------------------------------------------------
// TDM 2D tile load: Global -> LDS, f16 elements, row-major tile
// [tile_d1 rows x tile_d0 elems], source row stride stride0 elems.
// LDS padding: +4 dwords (8 halves) after each 32-dword (64-half) row
// => LDS row stride 72 halves (144 B, 16B-aligned rows).
// D# layout per CDNA5 ISA ch.8 (group0 128b, group1 256b; groups 2/3 zero).
// ---------------------------------------------------------------------------
static __device__ __forceinline__ void tdm_load_2d(unsigned int ldsoff,
                                                   const void* gaddr,
                                                   unsigned int tensor_d0,
                                                   unsigned int tensor_d1,
                                                   unsigned int tile_d0,
                                                   unsigned int tile_d1,
                                                   unsigned long long stride0) {
#if HAVE_TDM
  const unsigned long long ga = (unsigned long long)gaddr;
  u32x4 g0;
  g0[0] = 1u;                                               // count=1, user mode
  g0[1] = ldsoff;                                           // lds_addr
  g0[2] = (unsigned int)ga;                                 // global_addr[31:0]
  g0[3] = ((unsigned int)(ga >> 32) & 0x01FFFFFFu)          // global_addr[56:32]
        | 0x80000000u;                                      // type=2 ("image")
  i32x8 g1;
  g1[0] = (int)((1u << 16)    // data_size = 1 -> 2 bytes
              | (1u << 20)    // pad_enable
              | (4u << 22)    // pad_interval code 4 -> every 32 dwords
              | (3u << 25));  // pad_amount  code 3 -> 4 dwords
  g1[1] = (int)((tensor_d0 & 0xFFFFu) << 16);
  g1[2] = (int)(((tensor_d0 >> 16) & 0xFFFFu) | ((tensor_d1 & 0xFFFFu) << 16));
  g1[3] = (int)(((tensor_d1 >> 16) & 0xFFFFu) | ((tile_d0 & 0xFFFFu) << 16));
  g1[4] = (int)(tile_d1 & 0xFFFFu);                         // tile_dim1, tile_dim2=0
  g1[5] = (int)(unsigned int)stride0;                       // tensor_dim0_stride lo
  g1[6] = (int)((stride0 >> 32) & 0xFFFFu);                 // stride0 hi, stride1=0
  g1[7] = 0;
  const i32x4 z4 = {0, 0, 0, 0};
  const i32x8 z8 = {0, 0, 0, 0, 0, 0, 0, 0};
  // 6-arg form (clang-23 / therock-10.0 headers): groups 2,3 + extra group, cpol.
  __builtin_amdgcn_tensor_load_to_lds(g0, g1, z4, z4, z8, 0);
#endif
}

// ---------------------------------------------------------------------------
// Kernel 1: QKV projection (unchanged — good codegen).
// ---------------------------------------------------------------------------
__global__ __launch_bounds__(128)
void qkv_proj_kernel(const float* __restrict__ x,
                     const float* __restrict__ Wq,
                     const float* __restrict__ Wk,
                     const float* __restrict__ Wv,
                     _Float16* __restrict__ Qh,
                     _Float16* __restrict__ Kh,
                     _Float16* __restrict__ Vt) {
  const int lane  = threadIdx.x & 31;
  const int wave  = threadIdx.x >> 5;
  const int m0    = blockIdx.x * 128 + wave * 32;
  const int n0    = blockIdx.y * 64;
  const int which = blockIdx.z;
  const float* W  = (which == 0) ? Wq : (which == 1) ? Wk : Wv;

  const int l15   = lane & 15;
  const int half  = lane >> 4;
  const int koffA = half * 8;
  const int koffB = half * 8;

  const float* arow0 = x + (size_t)(m0 + l15) * D_MODEL;
  const float* arow1 = arow0 + (size_t)16 * D_MODEL;
  const float* brow  = W + (size_t)(n0 + l15) * D_MODEL + koffB;

  Raw16 ra0, ra1, rb[4];
  ldA_raw(arow0, 0, koffA, ra0);
  ldA_raw(arow1, 0, koffA, ra1);
#pragma unroll
  for (int j = 0; j < 4; ++j) ldB_raw(brow + (size_t)j * 16 * D_MODEL, rb[j]);

  v8f acc[2][4] = {};

  for (int k0 = 0; k0 < D_MODEL - 32; k0 += 32) {
    v16h a0 = cvt_frag(ra0), a1 = cvt_frag(ra1);
    v16h bf[4];
#pragma unroll
    for (int j = 0; j < 4; ++j) bf[j] = cvt_frag(rb[j]);

    const int kn = k0 + 32;
    ldA_raw(arow0, kn, koffA, ra0);
    ldA_raw(arow1, kn, koffA, ra1);
#pragma unroll
    for (int j = 0; j < 4; ++j) ldB_raw(brow + kn + (size_t)j * 16 * D_MODEL, rb[j]);

#pragma unroll
    for (int j = 0; j < 4; ++j) {
      acc[0][j] = wmma_f16(a0, bf[j], acc[0][j]);
      acc[1][j] = wmma_f16(a1, bf[j], acc[1][j]);
    }
  }
  {
    v16h a0 = cvt_frag(ra0), a1 = cvt_frag(ra1);
    v16h bf[4];
#pragma unroll
    for (int j = 0; j < 4; ++j) bf[j] = cvt_frag(rb[j]);
#pragma unroll
    for (int j = 0; j < 4; ++j) {
      acc[0][j] = wmma_f16(a0, bf[j], acc[0][j]);
      acc[1][j] = wmma_f16(a1, bf[j], acc[1][j]);
    }
  }

  const float scale = (which == 0) ? 0.125f : 1.0f;
#pragma unroll
  for (int i = 0; i < 2; ++i) {
    const int mbase = m0 + i * 16 + (half << 3);
#pragma unroll
    for (int r = 0; r < 8; ++r) {
      const int mm = mbase + r;
      const int bb = mm >> 11;
      const int ss = mm & (SEQ - 1);
#pragma unroll
      for (int ns = 0; ns < 4; ++ns) {
        const int nn = n0 + ns * 16 + l15;
        const int hh = nn >> 6;
        const int dd = nn & (HEAD_DIM - 1);
        const float val = acc[i][ns][r] * scale;
        if (which == 2) {
          Vt[(((size_t)bb * N_HEADS + hh) * HEAD_DIM + dd) * SEQ + ss] = (_Float16)val;
        } else {
          _Float16* dst = (which == 0) ? Qh : Kh;
          dst[(((size_t)bb * N_HEADS + hh) * SEQ + ss) * HEAD_DIM + dd] = (_Float16)val;
        }
      }
    }
  }
}

// ---------------------------------------------------------------------------
// Kernel 2: fused flash attention; K/V tiles staged to LDS by the Tensor
// Data Mover (double-buffered), fragments read from LDS by all 4 waves.
// grid = (SEQ/64, B*N_HEADS), block = 128.
// ---------------------------------------------------------------------------
#define TPAD 72   // LDS row stride in halves (64 data + 8 pad from TDM)

__global__ __launch_bounds__(128)
void flash_attn_kernel(const _Float16* __restrict__ Qh,
                       const _Float16* __restrict__ Kh,
                       const _Float16* __restrict__ Vt,
                       float* __restrict__ out) {
  __shared__ _Float16 Ktile[2][64][TPAD];  // [key][d]
  __shared__ _Float16 Vtile[2][64][TPAD];  // [d][key]
  __shared__ _Float16 Plds[4][16][TPAD];

  const int lane = threadIdx.x & 31;
  const int wave = threadIdx.x >> 5;
  const int bh   = blockIdx.y;
  const int q0   = blockIdx.x * 64 + wave * 16;

  const _Float16* Qb = Qh + (size_t)bh * SEQ * HEAD_DIM;
  const _Float16* Kb = Kh + (size_t)bh * SEQ * HEAD_DIM;
  const _Float16* Vb = Vt + (size_t)bh * HEAD_DIM * SEQ;   // [Hd][S]

  const int half = lane >> 4;
  const int l15  = lane & 15;
  const int koff = half * 8;
  const int doff = half * 8;

#if HAVE_TDM
  if (wave == 0) {   // TDM ignores EXEC; one issue per wave-0 instruction
    tdm_load_2d(lds_off(&Ktile[0][0][0]), Kb, HEAD_DIM, SEQ, 64, 64, HEAD_DIM);
    tdm_load_2d(lds_off(&Vtile[0][0][0]), Vb, SEQ, HEAD_DIM, 64, 64, SEQ);
  }
#else
  {  // cooperative fallback stage of tile 0
    const int row = threadIdx.x >> 1;
    const int c   = (threadIdx.x & 1) * 32;
#pragma unroll
    for (int j = 0; j < 4; ++j) {
      *(v8h*)&Ktile[0][row][c + j * 8] = *(const v8h*)(Kb + (size_t)row * HEAD_DIM + c + j * 8);
      *(v8h*)&Vtile[0][row][c + j * 8] = *(const v8h*)(Vb + (size_t)row * SEQ + c + j * 8);
    }
  }
#endif

  const _Float16* qrow = Qb + (size_t)(q0 + l15) * HEAD_DIM;
  const v16h qa0 = loadA_f16(qrow, koff);
  const v16h qa1 = loadA_f16(qrow + 32, koff);

  v8f o[4] = {v8f{}, v8f{}, v8f{}, v8f{}};
  float mrow[8], lrow[8];
#pragma unroll
  for (int r = 0; r < 8; ++r) { mrow[r] = -1e30f; lrow[r] = 0.0f; }

  for (int kb = 0; kb < SEQ; kb += 64) {
    const int buf = (kb >> 6) & 1;

#if HAVE_TDM
    if (wave == 0) __builtin_amdgcn_s_wait_tensorcnt(0);   // tile `buf` landed
#endif
    __syncthreads();   // tile ready for all; previous tile's readers done

#if HAVE_TDM
    if (wave == 0 && kb + 64 < SEQ) {   // overlap next DMA with compute
      tdm_load_2d(lds_off(&Ktile[buf ^ 1][0][0]),
                  Kb + (size_t)(kb + 64) * HEAD_DIM, HEAD_DIM, SEQ, 64, 64, HEAD_DIM);
      tdm_load_2d(lds_off(&Vtile[buf ^ 1][0][0]),
                  Vb + (size_t)(kb + 64), SEQ, HEAD_DIM, 64, 64, SEQ);
    }
#else
    if (kb + 64 < SEQ) {
      const int row = threadIdx.x >> 1;
      const int c   = (threadIdx.x & 1) * 32;
#pragma unroll
      for (int j = 0; j < 4; ++j) {
        *(v8h*)&Ktile[buf ^ 1][row][c + j * 8] =
            *(const v8h*)(Kb + (size_t)(kb + 64 + row) * HEAD_DIM + c + j * 8);
        *(v8h*)&Vtile[buf ^ 1][row][c + j * 8] =
            *(const v8h*)(Vb + (size_t)row * SEQ + kb + 64 + c + j * 8);
      }
    }
#endif

    // ---- scores S = (Q*scale) @ K^T : K fragments from LDS ----
    v8f s[4];
#pragma unroll
    for (int ns = 0; ns < 4; ++ns) {
      const _Float16* kp = &Ktile[buf][ns * 16 + l15][doff];
      v8f z = {};
      z = wmma_f16(qa0, loadB_f16(kp), z);
      z = wmma_f16(qa1, loadB_f16(kp + 32), z);
      s[ns] = z;
    }

    // ---- online softmax ----
#pragma unroll
    for (int r = 0; r < 8; ++r) {
      float mx = fmaxf(fmaxf(s[0][r], s[1][r]), fmaxf(s[2][r], s[3][r]));
      mx = fmaxf(mx, __shfl_xor(mx, 1));
      mx = fmaxf(mx, __shfl_xor(mx, 2));
      mx = fmaxf(mx, __shfl_xor(mx, 4));
      mx = fmaxf(mx, __shfl_xor(mx, 8));
      const float mnew = fmaxf(mrow[r], mx);
      const float corr = __expf(mrow[r] - mnew);
      mrow[r] = mnew;

      const float p0 = __expf(s[0][r] - mnew);
      const float p1 = __expf(s[1][r] - mnew);
      const float p2 = __expf(s[2][r] - mnew);
      const float p3 = __expf(s[3][r] - mnew);

      float psum = (p0 + p1) + (p2 + p3);
      psum += __shfl_xor(psum, 1);
      psum += __shfl_xor(psum, 2);
      psum += __shfl_xor(psum, 4);
      psum += __shfl_xor(psum, 8);
      lrow[r] = lrow[r] * corr + psum;

      o[0][r] *= corr; o[1][r] *= corr; o[2][r] *= corr; o[3][r] *= corr;

      const int prow = r + (half << 3);
      Plds[wave][prow][ 0 + l15] = (_Float16)p0;
      Plds[wave][prow][16 + l15] = (_Float16)p1;
      Plds[wave][prow][32 + l15] = (_Float16)p2;
      Plds[wave][prow][48 + l15] = (_Float16)p3;
    }

    // Wave-private LDS RAW on the P tile.
    asm volatile("s_wait_dscnt 0" ::: "memory");

    const _Float16* pp = &Plds[wave][l15][0];
    V16H up0, up1;
    up0.h[0] = *(const v8h*)(pp + koff);
    up0.h[1] = *(const v8h*)(pp + 16 + koff);
    up1.h[0] = *(const v8h*)(pp + 32 + koff);
    up1.h[1] = *(const v8h*)(pp + 48 + koff);

    // ---- O += P @ V : V fragments from LDS ----
#pragma unroll
    for (int dsub = 0; dsub < 4; ++dsub) {
      const _Float16* vp = &Vtile[buf][dsub * 16 + l15][doff];
      o[dsub] = wmma_f16(up0.v, loadB_f16(vp), o[dsub]);
      o[dsub] = wmma_f16(up1.v, loadB_f16(vp + 32), o[dsub]);
    }
  }

  // ---- normalize, write fp32 output [B, S, H*Hd] ----
  const int b = bh >> 4;
  const int h = bh & (N_HEADS - 1);
#pragma unroll
  for (int r = 0; r < 8; ++r) {
    const float inv = 1.0f / lrow[r];
    const int mm = q0 + r + (half << 3);
    float* orow = out + ((size_t)b * SEQ + mm) * D_MODEL + h * HEAD_DIM;
#pragma unroll
    for (int dsub = 0; dsub < 4; ++dsub)
      orow[dsub * 16 + l15] = o[dsub][r] * inv;
  }
}

// ---------------------------------------------------------------------------
extern "C" void kernel_launch(void* const* d_in, const int* in_sizes, int n_in,
                              void* d_out, int out_size, void* d_ws, size_t ws_size,
                              hipStream_t stream) {
  const float* x  = (const float*)d_in[0];
  const float* Wq = (const float*)d_in[1];
  const float* Wk = (const float*)d_in[2];
  const float* Wv = (const float*)d_in[3];
  float* out = (float*)d_out;

  const size_t elems = (size_t)BATCH * N_HEADS * SEQ * HEAD_DIM;
  _Float16* Qh = (_Float16*)d_ws;
  _Float16* Kh = Qh + elems;
  _Float16* Vt = Kh + elems;

  dim3 g1((BATCH * SEQ) / 128, D_MODEL / 64, 3);
  qkv_proj_kernel<<<g1, dim3(128), 0, stream>>>(x, Wq, Wk, Wv, Qh, Kh, Vt);

  dim3 g2(SEQ / 64, BATCH * N_HEADS);
  flash_attn_kernel<<<g2, dim3(128), 0, stream>>>(Qh, Kh, Vt, out);
}
